// FullLSTMQK_78460462563652
// MI455X (gfx1250) — matrix operand order repaired and checked by reading, not compiled, not run on hardware
//
#include <hip/hip_runtime.h>
#include <hip/hip_bf16.h>

// ---------------- problem constants ----------------
#define BB   8
#define SS   2048
#define DD   1024
#define G4D  4096          // 4*D gate width
#define KTOT 2048          // concat [h ; x] K dimension
#define OHH  64
#define QDD  1024
#define KDD  256

typedef __attribute__((ext_vector_type(16))) __bf16 v16bf;
typedef __attribute__((ext_vector_type(8)))  __bf16 v8bf;
typedef __attribute__((ext_vector_type(8)))  float  v8f;

typedef __attribute__((address_space(3))) char lds_char_t;
static __device__ inline unsigned lds_off_u32(const void* p) {
  return (unsigned)(unsigned long long)(lds_char_t*)p;   // addrspacecast -> LDS byte offset
}

// A-operand loader for V_WMMA_*_16X16X32_BF16 (branch-free).
// lane_base must already point at this lane's row (row m = lane&15).
// half = lane>>4 selects K groups: elems 0..7 -> K = kbase+8*half+0..7,
//                                  elems 8..15 -> K = kbase+16+8*half+0..7.
static __device__ inline v16bf load_a_kb(const __bf16* lane_base, int kbase, int half) {
  const __bf16* p = lane_base + kbase + 8 * half;
  v8bf lo = *(const v8bf*)(p);
  v8bf hi = *(const v8bf*)(p + 16);
  return __builtin_shufflevector(lo, hi, 0,1,2,3,4,5,6,7,8,9,10,11,12,13,14,15);
}

// B-operand loader: B tile is K-major in memory (row k contiguous over n);
// lane l holds column N = l&15, elems 0..7 -> K = krow+8*(l>>4)+0..7,
// elems 8..15 -> K = krow+16+8*(l>>4)+0..7.
static __device__ inline v16bf load_b_tile(const __bf16* __restrict__ base,
                                           long row_stride, int krow, int n0,
                                           int lane) {
  const __bf16* p = base + (long)(krow + 8 * (lane >> 4)) * row_stride + n0 + (lane & 15);
  v16bf r;
#pragma unroll
  for (int s = 0; s < 8; ++s) {
    r[s] = p[(long)s * row_stride];
    r[s + 8] = p[(long)(s + 16) * row_stride];
  }
  return r;
}

static __device__ inline v8f zero_acc() {
  v8f a;
#pragma unroll
  for (int i = 0; i < 8; ++i) a[i] = 0.0f;
  return a;
}

static __device__ inline float sigmoidf_(float x) {
  return 1.0f / (1.0f + __expf(-x));
}

// ---------------- conversion / transpose kernels ----------------

__global__ void cvt_bf16_kernel(__bf16* __restrict__ dst,
                                const float* __restrict__ src, long n) {
  long stride = (long)gridDim.x * blockDim.x;
  for (long i = (long)blockIdx.x * blockDim.x + threadIdx.x; i < n; i += stride)
    dst[i] = (__bf16)src[i];
}

// dst (K x N) bf16  <-  transpose of src (N x K) f32 : dst[k*N+n] = src[n*K+k]
__global__ void transpose_bf16_kernel(__bf16* __restrict__ dst,
                                      const float* __restrict__ src,
                                      int N, int K, long total) {
  long stride = (long)gridDim.x * blockDim.x;
  for (long i = (long)blockIdx.x * blockDim.x + threadIdx.x; i < total; i += stride) {
    long k = i / N;
    long n = i - k * (long)N;
    dst[i] = (__bf16)src[n * (long)K + k];
  }
}

// ---------------- persistent LSTM recurrence ----------------
// grid.x = 128 : blocks 0..63 -> q LSTM, 64..127 -> k LSTM.
// Each block owns 16 hidden units. 4 waves = one gate each.
// Per step: C(16x16) = A(16x32)·B(32x16) over 64 K-tiles,
// A = [h_{t-1} ; x_t]; batch rows 0..7 real, 8..15 zero
// (hbuf rows 8..15 pre-zeroed; x pad lanes read a zeroed LDS row).
// x_{t+1} is prefetched into LDS via CDNA5 async global->LDS copies.
__global__ __launch_bounds__(128)
void lstm_seq_kernel(const __bf16* __restrict__ xb,      // (B,S,D) bf16
                     __bf16* __restrict__ hbuf,          // [2 lstm][2 buf][16*D] (rows 8..15 zero)
                     __bf16* __restrict__ hsq,           // (B,S,D) bf16
                     __bf16* __restrict__ hsk,
                     const __bf16* __restrict__ WTq,     // (2048 x 4096) bf16 K-major
                     const __bf16* __restrict__ WTk,
                     const float* __restrict__ bi_q, const float* __restrict__ bh_q,
                     const float* __restrict__ bi_k, const float* __restrict__ bh_k,
                     const float* __restrict__ cq_in, const float* __restrict__ ck_in,
                     float* __restrict__ out_states,     // hq|cq|hk|ck (fp32)
                     unsigned* __restrict__ bar) {
  __shared__ __bf16 xlds[2][BB * DD];   // double-buffered x_t tile (8 x 1024)
  __shared__ __bf16 zrow[DD + 32];      // zero row for M-padding lanes
  __shared__ float  glds[4 * 128];      // [gate][m=0..7][j=0..15]

  const int tid  = threadIdx.x;
  const int lane = tid & 31;
  const int gate = tid >> 5;                 // wave id = gate {i,f,g,o}
  const int lstm  = (int)blockIdx.x >> 6;
  const int slice = (int)blockIdx.x & 63;
  const int j0 = slice << 4;

  const __bf16* WT  = lstm ? WTk  : WTq;
  const float*  bi  = lstm ? bi_k : bi_q;
  const float*  bh  = lstm ? bh_k : bh_q;
  const float*  cin = lstm ? ck_in : cq_in;
  __bf16* hs = lstm ? hsk : hsq;
  __bf16* hbq = hbuf + (long)lstm * 2 * 16 * DD;   // two ping-pong buffers, 16 rows each
  unsigned* cnt = bar + lstm * 2;
  unsigned* gen = cnt + 1;

  const int nb = (gate << 10) + j0;                   // this wave's N column base
  const float bias = bi[nb + (lane & 15)] + bh[nb + (lane & 15)];

  const int m    = lane & 15;
  const int half = lane >> 4;
  const int mc   = m < 8 ? m : 7;

  // pointwise ownership: thread -> (batch, hidden-within-slice)
  const int pb = tid >> 4;
  const int pj = tid & 15;
  float creg = cin[pb * DD + j0 + pj];

  // zero the pad row
  for (int i = tid; i < DD + 32; i += 128) zrow[i] = (__bf16)0.0f;
  __syncthreads();

  // prologue: async-stage x_0 into LDS buffer 0
  {
    const __bf16* gsrc = xb;  // t = 0
#pragma unroll
    for (int i = 0; i < 8; ++i) {
      int c = tid + (i << 7);            // 16B chunk id, 0..1023
      int mr = c >> 6;                   // batch row
      int col = (c & 63) << 3;           // element within row
      unsigned loff = lds_off_u32(&xlds[0][mr * DD + col]);
      unsigned long long ga = (unsigned long long)(gsrc + (long)mr * SS * DD + col);
      asm volatile("global_load_async_to_lds_b128 %0, %1, off"
                   :: "v"(loff), "v"(ga) : "memory");
    }
  }

  int p = 0;
  for (int t = 0; t < SS; ++t) {
    // wait for this wave's async x stage, then make it visible block-wide
    asm volatile("s_wait_asynccnt 0x0" ::: "memory");
    __syncthreads();

    const __bf16* hcur = hbq + p * (16 * DD);
    const __bf16* hab  = hcur + (long)m * DD;                      // rows 8..15 are zero
    const __bf16* xab  = (m < 8) ? &xlds[t & 1][mc * DD] : &zrow[0];

    v8f acc0 = zero_acc();
    v8f acc1 = zero_acc();

    // h part: K tiles 0..31 (A from global hbuf broadcast, B from L2-resident WT)
#pragma unroll 4
    for (int kt = 0; kt < 32; kt += 2) {
      v16bf a0 = load_a_kb(hab, kt * 32, half);
      v16bf b0 = load_b_tile(WT, G4D, kt * 32, nb, lane);
      v16bf a1 = load_a_kb(hab, kt * 32 + 32, half);
      v16bf b1 = load_b_tile(WT, G4D, kt * 32 + 32, nb, lane);
      __builtin_prefetch((const void*)(WT + ((long)(kt * 32 + lane) + 64) * G4D + nb), 0, 1);
      acc0 = __builtin_amdgcn_wmma_f32_16x16x32_bf16(false, a0, false, b0,
                                                     (short)0, acc0, false, false);
      acc1 = __builtin_amdgcn_wmma_f32_16x16x32_bf16(false, a1, false, b1,
                                                     (short)0, acc1, false, false);
    }
    // x part: K tiles 32..63 (A from LDS-staged x_t, B from WT)
#pragma unroll 4
    for (int kt = 0; kt < 32; kt += 2) {
      v16bf a0 = load_a_kb(xab, kt * 32, half);
      v16bf b0 = load_b_tile(WT, G4D, 1024 + kt * 32, nb, lane);
      v16bf a1 = load_a_kb(xab, kt * 32 + 32, half);
      v16bf b1 = load_b_tile(WT, G4D, 1024 + kt * 32 + 32, nb, lane);
      if (kt < 30)
        __builtin_prefetch((const void*)(WT + ((long)(1024 + kt * 32 + lane) + 64) * G4D + nb), 0, 1);
      acc0 = __builtin_amdgcn_wmma_f32_16x16x32_bf16(false, a0, false, b0,
                                                     (short)0, acc0, false, false);
      acc1 = __builtin_amdgcn_wmma_f32_16x16x32_bf16(false, a1, false, b1,
                                                     (short)0, acc1, false, false);
    }

    // kick off async stage of x_{t+1} while we finish this step
    if (t + 1 < SS) {
      const __bf16* gsrc = xb + (long)(t + 1) * DD;
      const int nbuf = (t + 1) & 1;
#pragma unroll
      for (int i = 0; i < 8; ++i) {
        int c = tid + (i << 7);
        int mr = c >> 6;
        int col = (c & 63) << 3;
        unsigned loff = lds_off_u32(&xlds[nbuf][mr * DD + col]);
        unsigned long long ga = (unsigned long long)(gsrc + (long)mr * SS * DD + col);
        asm volatile("global_load_async_to_lds_b128 %0, %1, off"
                     :: "v"(loff), "v"(ga) : "memory");
      }
    }

    // C layout: vgpr r, lane l -> M = r + 8*(l>>4), N = l&15.
    // Batch rows 0..7 live in lanes 0..15, vgpr 0..7.
    if (lane < 16) {
#pragma unroll
      for (int r = 0; r < 8; ++r)
        glds[(gate << 7) + (r << 4) + lane] = acc0[r] + acc1[r] + bias;
    }
    __syncthreads();

    // pointwise LSTM cell update (one thread per (batch, hidden))
    {
      float gi = glds[0 * 128 + tid];
      float gf = glds[1 * 128 + tid];
      float gg = glds[2 * 128 + tid];
      float go = glds[3 * 128 + tid];
      creg = sigmoidf_(gf) * creg + sigmoidf_(gi) * tanhf(gg);
      float h = sigmoidf_(go) * tanhf(creg);
      __bf16 h16 = (__bf16)h;
      hbq[(1 - p) * (16 * DD) + pb * DD + j0 + pj] = h16;               // broadcast
      hs[(long)pb * SS * DD + (long)t * DD + j0 + pj] = h16;            // sequence
      if (t == SS - 1) {
        float* oh = out_states + (long)lstm * 2 * BB * DD;
        oh[pb * DD + j0 + pj] = h;                  // final h
        oh[BB * DD + pb * DD + j0 + pj] = creg;     // final c
      }
    }

    // device-scope barrier across the 64 workgroups of this LSTM
    __threadfence();
    __syncthreads();
    if (tid == 0) {
      unsigned g = __hip_atomic_load(gen, __ATOMIC_ACQUIRE, __HIP_MEMORY_SCOPE_AGENT);
      unsigned old = __hip_atomic_fetch_add(cnt, 1u, __ATOMIC_ACQ_REL, __HIP_MEMORY_SCOPE_AGENT);
      if (old == 63u) {
        __hip_atomic_store(cnt, 0u, __ATOMIC_RELAXED, __HIP_MEMORY_SCOPE_AGENT);
        __hip_atomic_store(gen, g + 1u, __ATOMIC_RELEASE, __HIP_MEMORY_SCOPE_AGENT);
      } else {
        while (__hip_atomic_load(gen, __ATOMIC_ACQUIRE, __HIP_MEMORY_SCOPE_AGENT) == g)
          __builtin_amdgcn_s_sleep(2);
      }
    }
    __syncthreads();
    p ^= 1;
  }
}

// ---------------- fused output MLP: relu(h@W1T+b1)@W2T+b2 ----------------
// grid = (512, 2) : x = 32-row block of (B*S), y = 0 (q) / 1 (k). 4 waves.
__global__ __launch_bounds__(128)
void mlp_kernel(const __bf16* __restrict__ hsq, const __bf16* __restrict__ hsk,
                const __bf16* __restrict__ W1Tq, const __bf16* __restrict__ W2Tq,
                const __bf16* __restrict__ W1Tk, const __bf16* __restrict__ W2Tk,
                const float* __restrict__ b1q, const float* __restrict__ b2q,
                const float* __restrict__ b1k, const float* __restrict__ b2k,
                float* __restrict__ out) {
  __shared__ __bf16 hid[32 * OHH];   // relu hidden, (32 rows x 64) row-major

  const int tid = threadIdx.x, lane = tid & 31, wv = tid >> 5;
  const int isK = blockIdx.y;
  const __bf16* hs  = isK ? hsk  : hsq;
  const __bf16* W1T = isK ? W1Tk : W1Tq;   // (1024 x 64)  K-major
  const __bf16* W2T = isK ? W2Tk : W2Tq;   // (64 x ND)    K-major
  const float* b1 = isK ? b1k : b1q;
  const float* b2 = isK ? b2k : b2q;
  float* op = out + (isK ? (long)BB * SS * QDD : 0);
  const int ND = isK ? KDD : QDD;

  const long r0 = (long)blockIdx.x * 32;
  const int mt = wv & 1;       // which 16-row half
  const int ng = wv >> 1;      // which pair of hidden N-tiles
  const int half = lane >> 4, nl = lane & 15;
  const int m = lane & 15;

  // GEMM1: (32x1024)·(1024x64), each wave: 1 M-tile x 2 N-tiles, K loop 32
  v8f acc0 = zero_acc(), acc1 = zero_acc();
  const __bf16* Abase = hs + (r0 + mt * 16 + m) * DD;   // per-lane row pointer
#pragma unroll 4
  for (int kt = 0; kt < 32; ++kt) {
    v16bf a  = load_a_kb(Abase, kt * 32, half);
    v16bf b0 = load_b_tile(W1T, OHH, kt * 32, (ng * 2 + 0) * 16, lane);
    v16bf b1v = load_b_tile(W1T, OHH, kt * 32, (ng * 2 + 1) * 16, lane);
    acc0 = __builtin_amdgcn_wmma_f32_16x16x32_bf16(false, a, false, b0,
                                                   (short)0, acc0, false, false);
    acc1 = __builtin_amdgcn_wmma_f32_16x16x32_bf16(false, a, false, b1v,
                                                   (short)0, acc1, false, false);
  }
#pragma unroll
  for (int j = 0; j < 2; ++j) {
    int n0 = (ng * 2 + j) * 16;
    float bs = b1[n0 + nl];
    v8f A = j ? acc1 : acc0;
#pragma unroll
    for (int r = 0; r < 8; ++r) {
      float v = A[r] + bs;
      v = v > 0.0f ? v : 0.0f;
      hid[(mt * 16 + r + 8 * half) * OHH + n0 + nl] = (__bf16)v;
    }
  }
  __syncthreads();

  // GEMM2: (32x64)·(64xND), each wave: 1 M-tile x (ND/16/2) N-tiles, K = 2 tiles
  const __bf16* hlane = &hid[(mt * 16 + m) * OHH];
  v16bf a0 = load_a_kb(hlane, 0, half);
  v16bf a1 = load_a_kb(hlane, 32, half);
  const int ntHalf = (ND / 16) / 2;
  for (int nt = ng * ntHalf; nt < (ng + 1) * ntHalf; ++nt) {
    v8f acc = zero_acc();
    v16bf b0 = load_b_tile(W2T, ND, 0, nt * 16, lane);
    v16bf b1v = load_b_tile(W2T, ND, 32, nt * 16, lane);
    acc = __builtin_amdgcn_wmma_f32_16x16x32_bf16(false, a0, false, b0,
                                                  (short)0, acc, false, false);
    acc = __builtin_amdgcn_wmma_f32_16x16x32_bf16(false, a1, false, b1v,
                                                  (short)0, acc, false, false);
    const int col = nt * 16 + nl;
    const float bb = b2[col];
#pragma unroll
    for (int r = 0; r < 8; ++r) {
      long row = r0 + mt * 16 + r + 8 * half;
      op[row * ND + col] = acc[r] + bb;
    }
  }
}

// ---------------- host driver ----------------
extern "C" void kernel_launch(void* const* d_in, const int* in_sizes, int n_in,
                              void* d_out, int out_size, void* d_ws, size_t ws_size,
                              hipStream_t stream) {
  (void)in_sizes; (void)n_in; (void)out_size; (void)ws_size;
  const float* x    = (const float*)d_in[0];
  const float* hq   = (const float*)d_in[1];
  const float* cq   = (const float*)d_in[2];
  const float* hk   = (const float*)d_in[3];
  const float* ck   = (const float*)d_in[4];
  const float* Wi_q = (const float*)d_in[5];
  const float* Wh_q = (const float*)d_in[6];
  const float* bi_q = (const float*)d_in[7];
  const float* bh_q = (const float*)d_in[8];
  const float* Wi_k = (const float*)d_in[9];
  const float* Wh_k = (const float*)d_in[10];
  const float* bi_k = (const float*)d_in[11];
  const float* bh_k = (const float*)d_in[12];
  const float* W1_q = (const float*)d_in[13];
  const float* b1_q = (const float*)d_in[14];
  const float* W2_q = (const float*)d_in[15];
  const float* b2_q = (const float*)d_in[16];
  const float* W1_k = (const float*)d_in[17];
  const float* b1_k = (const float*)d_in[18];
  const float* W2_k = (const float*)d_in[19];
  const float* b2_k = (const float*)d_in[20];

  char* ws = (char*)d_ws;
  const long NX = (long)BB * SS * DD;          // 16,777,216
  long off = 0;
  __bf16* xb   = (__bf16*)(ws + off); off += NX * 2;                  // x bf16
  __bf16* hbuf = (__bf16*)(ws + off); off += 2L * 2 * 16 * DD * 2;    // h ping-pong (16 rows)
  __bf16* hsq  = (__bf16*)(ws + off); off += NX * 2;                  // hq sequence
  __bf16* hsk  = (__bf16*)(ws + off); off += NX * 2;                  // hk sequence
  __bf16* WTq  = (__bf16*)(ws + off); off += (long)KTOT * G4D * 2;    // [WhT;WiT] q
  __bf16* WTk  = (__bf16*)(ws + off); off += (long)KTOT * G4D * 2;
  __bf16* W1Tq = (__bf16*)(ws + off); off += (long)DD * OHH * 2;
  __bf16* W2Tq = (__bf16*)(ws + off); off += (long)OHH * QDD * 2;
  __bf16* W1Tk = (__bf16*)(ws + off); off += (long)DD * OHH * 2;
  __bf16* W2Tk = (__bf16*)(ws + off); off += (long)OHH * KDD * 2;
  unsigned* bar = (unsigned*)(ws + off); off += 256;

  float* out = (float*)d_out;
  float* out_states = out + (long)BB * SS * QDD + (long)BB * SS * KDD;

  hipMemsetAsync(bar, 0, 256, stream);
  hipMemsetAsync(hbuf, 0, 2L * 2 * 16 * DD * 2, stream);  // pad rows 8..15 stay zero

  // one-time conversions
  cvt_bf16_kernel<<<4096, 256, 0, stream>>>(xb, x, NX);
  cvt_bf16_kernel<<<32, 256, 0, stream>>>(hbuf, hq, (long)BB * DD);                     // q buf0 rows 0..7
  cvt_bf16_kernel<<<32, 256, 0, stream>>>(hbuf + 2L * 16 * DD, hk, (long)BB * DD);      // k buf0 rows 0..7
  // WT = [Wh^T ; Wi^T] : (2048 x 4096) bf16, K-major
  transpose_bf16_kernel<<<4096, 256, 0, stream>>>(WTq, Wh_q, G4D, DD, (long)DD * G4D);
  transpose_bf16_kernel<<<4096, 256, 0, stream>>>(WTq + (long)DD * G4D, Wi_q, G4D, DD, (long)DD * G4D);
  transpose_bf16_kernel<<<4096, 256, 0, stream>>>(WTk, Wh_k, G4D, DD, (long)DD * G4D);
  transpose_bf16_kernel<<<4096, 256, 0, stream>>>(WTk + (long)DD * G4D, Wi_k, G4D, DD, (long)DD * G4D);
  transpose_bf16_kernel<<<256, 256, 0, stream>>>(W1Tq, W1_q, OHH, DD, (long)DD * OHH);
  transpose_bf16_kernel<<<256, 256, 0, stream>>>(W2Tq, W2_q, QDD, OHH, (long)OHH * QDD);
  transpose_bf16_kernel<<<256, 256, 0, stream>>>(W1Tk, W1_k, OHH, DD, (long)DD * OHH);
  transpose_bf16_kernel<<<64, 256, 0, stream>>>(W2Tk, W2_k, KDD, OHH, (long)OHH * KDD);

  // persistent recurrence: 128 WGs (64 per LSTM), 4 waves each
  lstm_seq_kernel<<<128, 128, 0, stream>>>(xb, hbuf, hsq, hsk, WTq, WTk,
                                           bi_q, bh_q, bi_k, bh_k, cq, ck,
                                           out_states, bar);

  // fused MLP heads
  mlp_kernel<<<dim3(512, 2), 128, 0, stream>>>(hsq, hsk, W1Tq, W2Tq, W1Tk, W2Tk,
                                               b1_q, b2_q, b1_k, b2_k, out);
}
